// RGAT_13735305413409
// MI455X (gfx1250) — matrix operand order repaired and checked by reading, not compiled, not run on hardware
//
#include <hip/hip_runtime.h>

// ---- problem constants (match reference) ----
#define NN   30000
#define EE   480000
#define RR   8
#define BB   4
#define IND  128
#define H1D  256
#define OUTD 64
#define NEG_SLOPE 0.2f
#define LDS_PAD 16          // halves of padding per B-tile row (8 dwords)

typedef _Float16 v16h __attribute__((ext_vector_type(16)));
typedef _Float16 v8h  __attribute__((ext_vector_type(8)));
typedef float    v8f  __attribute__((ext_vector_type(8)));
typedef unsigned int u32x4 __attribute__((ext_vector_type(4)));
typedef int          i32x4 __attribute__((ext_vector_type(4)));
typedef int          i32x8 __attribute__((ext_vector_type(8)));

// float atomic max via sign-aware integer atomics (works with 0xFF..F init)
__device__ __forceinline__ void atomicMaxF(float* addr, float v) {
  if (v >= 0.0f) atomicMax((int*)addr, __float_as_int(v));
  else           atomicMin((unsigned int*)addr, __float_as_uint(v));
}

// ---------------------------------------------------------------------------
// W[r] = sum_b comp[r,b] * basis[b]  -> stored TRANSPOSED as Wt[r][o][k] (f16)
// so WMMA B fragments / TDM rows are contiguous in K.
// ---------------------------------------------------------------------------
__global__ __launch_bounds__(256)
void build_wt(const float* __restrict__ basis, const float* __restrict__ comp,
              _Float16* __restrict__ wt, int Kd, int HO) {
  int t = blockIdx.x * 256 + threadIdx.x;
  int total = RR * Kd * HO;
  if (t >= total) return;
  int r = t / (Kd * HO);
  int rem = t % (Kd * HO);
  int kk = rem / HO;
  int o  = rem % HO;
  float w = 0.f;
#pragma unroll
  for (int b = 0; b < BB; ++b)
    w += comp[r * BB + b] * basis[((size_t)b * Kd + kk) * HO + o];
  wt[((size_t)r * HO + o) * Kd + kk] = (_Float16)w;
}

__global__ __launch_bounds__(256)
void cvt_f16(const float* __restrict__ in, _Float16* __restrict__ out, int total) {
  int t = blockIdx.x * 256 + threadIdx.x;
  if (t < total) out[t] = (_Float16)in[t];
}

// ---------------------------------------------------------------------------
// Dense per-relation transform: XR[r][n][:] = Xf16[n][:] @ W[r]
// 8 waves per block; the 64-col x Kd B tile (relation weights, f16) is staged
// into LDS once per block by the Tensor Data Mover (tensor_load_to_lds) with
// per-row padding of 8 dwords so ds_load_b128 B-fragment reads are
// bank-conflict free. Each wave computes a 16x64 tile: A fragments stream
// from global (two 16B loads / K-step), B fragments from LDS, 4 accumulators
// share each A fragment via v_wmma_f32_16x16x32_f16 (f32 accumulate).
// ---------------------------------------------------------------------------
__global__ __launch_bounds__(256)
void rgat_gemm_wmma(const _Float16* __restrict__ X,   // [NN][Kd] row-major
                    const _Float16* __restrict__ Wt,  // [RR][HO][Kd]
                    float* __restrict__ XR,           // [RR][NN][HO]
                    int Kd, int HO) {
  __shared__ __attribute__((aligned(64))) _Float16 ldsB[64 * (256 + LDS_PAD)];
  const int lane = threadIdx.x & 31;
  const int wave = threadIdx.x >> 5;
  const int lrow = lane & 15;
  const int lhi  = lane >> 4;
  const int r  = blockIdx.y;
  const int c0 = blockIdx.z * 64;
  const int m0 = (blockIdx.x * 8 + wave) * 16;
  const int pitch = Kd + LDS_PAD;

  if (threadIdx.x < 32) {
    // --- TDM descriptor: 2-D tile, 64 rows x Kd halves of Wt[r] from row c0.
    unsigned long long gaddr =
        (unsigned long long)(Wt + ((size_t)r * HO + c0) * Kd);
    unsigned lds_off = (unsigned)(unsigned long long)(void*)ldsB; // low32 = LDS byte offset
    u32x4 g0;
    g0[0] = 1u;                                       // count = 1 (valid D#)
    g0[1] = lds_off;                                  // lds_addr
    g0[2] = (unsigned)(gaddr & 0xFFFFFFFFu);          // global_addr[31:0]
    g0[3] = (unsigned)((gaddr >> 32) & 0x01FFFFFFu)   // global_addr[56:32]
          | (2u << 30);                               // type = 2 ("image")
    unsigned pad_int = (Kd == 128) ? 5u : 6u;         // row = 64 / 128 dwords
    i32x8 g1;
    g1[0] = (int)((1u << 16)                          // data_size = 2 bytes
                | (1u << 20)                          // pad_enable
                | (pad_int << 22)                     // pad_interval
                | (7u << 25));                        // pad_amount = 8 dwords
    g1[1] = (int)(((unsigned)Kd & 0xFFFFu) << 16);    // tensor_dim0 lo16
    g1[2] = (int)(((unsigned)HO & 0xFFFFu) << 16);    // tensor_dim1 lo16
    g1[3] = (int)(((unsigned)Kd & 0xFFFFu) << 16);    // tile_dim0 = Kd
    g1[4] = 64;                                       // tile_dim1 = 64 rows
    g1[5] = Kd;                                       // tensor_dim0_stride lo32
    g1[6] = 0;
    g1[7] = 0;
    i32x4 gz4 = {0, 0, 0, 0};                         // 2-D: groups 2/3 unused
    i32x8 gz8 = {0, 0, 0, 0, 0, 0, 0, 0};
    __builtin_amdgcn_tensor_load_to_lds(g0, g1, gz4, gz4, gz8, 0);
    __builtin_amdgcn_s_wait_tensorcnt(0);
  }
  __syncthreads();

  if (m0 < NN) {
    const _Float16* xrow = X + (size_t)(m0 + lrow) * Kd;
    const _Float16* b0 = ldsB + (size_t)(lrow +  0) * pitch + lhi * 16;
    const _Float16* b1 = ldsB + (size_t)(lrow + 16) * pitch + lhi * 16;
    const _Float16* b2 = ldsB + (size_t)(lrow + 32) * pitch + lhi * 16;
    const _Float16* b3 = ldsB + (size_t)(lrow + 48) * pitch + lhi * 16;

    v8f acc0 = {}; v8f acc1 = {}; v8f acc2 = {}; v8f acc3 = {};

    for (int kb = 0; kb < Kd; kb += 32) {
      v8h alo = *(const v8h*)(xrow + kb + lhi * 16);
      v8h ahi = *(const v8h*)(xrow + kb + 8 + lhi * 16);
      v16h a;
#pragma unroll
      for (int i = 0; i < 8; ++i) { a[i] = alo[i]; a[8 + i] = ahi[i]; }
      v16h f0 = *(const v16h*)(b0 + kb);
      v16h f1 = *(const v16h*)(b1 + kb);
      v16h f2 = *(const v16h*)(b2 + kb);
      v16h f3 = *(const v16h*)(b3 + kb);
      acc0 = __builtin_amdgcn_wmma_f32_16x16x32_f16(false, a, false, f0, (short)0, acc0, false, false);
      acc1 = __builtin_amdgcn_wmma_f32_16x16x32_f16(false, a, false, f1, (short)0, acc1, false, false);
      acc2 = __builtin_amdgcn_wmma_f32_16x16x32_f16(false, a, false, f2, (short)0, acc2, false, false);
      acc3 = __builtin_amdgcn_wmma_f32_16x16x32_f16(false, a, false, f3, (short)0, acc3, false, false);
    }

    // C/D layout: VGPR i -> (M = i + 8*(lane>=16), N = lane&15)
    float* outbase = XR + ((size_t)r * NN + m0) * HO + c0;
#pragma unroll
    for (int i = 0; i < 8; ++i) {
      float* orow = outbase + (size_t)(i + lhi * 8) * HO + lrow;
      orow[0]  = acc0[i];
      orow[16] = acc1[i];
      orow[32] = acc2[i];
      orow[48] = acc3[i];
    }
  }
}

// ---------------------------------------------------------------------------
// Edge pipeline (f32)
// ---------------------------------------------------------------------------
__global__ __launch_bounds__(256)
void edge_logits(const float* __restrict__ XR, const int* __restrict__ ei,
                 const int* __restrict__ et, const float* __restrict__ q,
                 const float* __restrict__ kw, float* __restrict__ alog,
                 float* __restrict__ amax, int H, int O, int HO) {
  int t = blockIdx.x * 256 + threadIdx.x;
  if (t >= EE * H) return;
  int e = t / H, h = t % H;
  int s = ei[e], d = ei[EE + e], r = et[e];
  const float* xs = XR + ((size_t)r * NN + s) * HO + h * O;
  const float* xd = XR + ((size_t)r * NN + d) * HO + h * O;
  const float* qr = q  + (size_t)r * HO + h * O;
  const float* kr = kw + (size_t)r * HO + h * O;
  float acc = 0.f;
  for (int o = 0; o < O; ++o) acc += xd[o] * qr[o] + xs[o] * kr[o];
  float a = acc > 0.f ? acc : NEG_SLOPE * acc;
  alog[t] = a;
  atomicMaxF(&amax[(size_t)d * H + h], a);
}

__global__ __launch_bounds__(256)
void edge_exp(const int* __restrict__ ei, float* __restrict__ alog,
              const float* __restrict__ amax, float* __restrict__ denom, int H) {
  int t = blockIdx.x * 256 + threadIdx.x;
  if (t >= EE * H) return;
  int e = t / H, h = t % H;
  int d = ei[EE + e];
  float v = expf(alog[t] - amax[(size_t)d * H + h]);
  alog[t] = v;
  atomicAdd(&denom[(size_t)d * H + h], v);
}

__global__ __launch_bounds__(256)
void edge_scatter(const float* __restrict__ XR, const int* __restrict__ ei,
                  const int* __restrict__ et, const float* __restrict__ aexp,
                  const float* __restrict__ denom, float* __restrict__ agg,
                  int H, int O, int HO) {
  int t = blockIdx.x * 256 + threadIdx.x;
  if (t >= EE * HO) return;
  int e = t / HO, c = t % HO;
  int h = c / O;
  int s = ei[e], d = ei[EE + e], r = et[e];
  float coef = aexp[(size_t)e * H + h] / (denom[(size_t)d * H + h] + 1e-16f);
  float v = XR[((size_t)r * NN + s) * HO + c] * coef;
  atomicAdd(&agg[(size_t)d * HO + c], v);
}

__global__ __launch_bounds__(256)
void finish_relu_f16(const float* __restrict__ agg, const float* __restrict__ bias,
                     _Float16* __restrict__ out, int HO, int total) {
  int t = blockIdx.x * 256 + threadIdx.x;
  if (t >= total) return;
  float v = agg[t] + bias[t % HO];
  out[t] = (_Float16)(v > 0.f ? v : 0.f);
}

__global__ __launch_bounds__(256)
void finish_bias_f32(const float* __restrict__ agg, const float* __restrict__ bias,
                     float* __restrict__ out, int HO, int total) {
  int t = blockIdx.x * 256 + threadIdx.x;
  if (t >= total) return;
  out[t] = agg[t] + bias[t % HO];
}

// ---------------------------------------------------------------------------
extern "C" void kernel_launch(void* const* d_in, const int* in_sizes, int n_in,
                              void* d_out, int out_size, void* d_ws, size_t ws_size,
                              hipStream_t stream) {
  (void)in_sizes; (void)n_in; (void)out_size; (void)ws_size;
  const float* x      = (const float*)d_in[0];
  const int*   ei     = (const int*)d_in[1];
  const int*   et     = (const int*)d_in[2];
  const float* basis1 = (const float*)d_in[3];  const float* comp1 = (const float*)d_in[4];
  const float* q1     = (const float*)d_in[5];  const float* k1    = (const float*)d_in[6];
  const float* bias1  = (const float*)d_in[7];
  const float* basis2 = (const float*)d_in[8];  const float* comp2 = (const float*)d_in[9];
  const float* q2     = (const float*)d_in[10]; const float* k2    = (const float*)d_in[11];
  const float* bias2  = (const float*)d_in[12];
  const float* basis3 = (const float*)d_in[13]; const float* comp3 = (const float*)d_in[14];
  const float* q3     = (const float*)d_in[15]; const float* k3    = (const float*)d_in[16];
  const float* bias3  = (const float*)d_in[17];
  float* out = (float*)d_out;

  // workspace carve (256B aligned)
  char* p = (char*)d_ws;
  auto carve = [&](size_t bytes) -> void* {
    void* r = (void*)p; p += (bytes + 255) & ~(size_t)255; return r;
  };
  _Float16* wt1  = (_Float16*)carve((size_t)RR * IND * H1D * 2);
  _Float16* wt2  = (_Float16*)carve((size_t)RR * H1D * H1D * 2);
  _Float16* wt3  = (_Float16*)carve((size_t)RR * H1D * OUTD * 2);
  _Float16* xf   = (_Float16*)carve((size_t)NN * H1D * 2);
  float* xr      = (float*)carve((size_t)RR * NN * H1D * 4);
  float* alog    = (float*)carve((size_t)EE * 4 * 4);
  float* amax    = (float*)carve((size_t)NN * 4 * 4);
  float* denom   = (float*)carve((size_t)NN * 4 * 4);
  float* agg     = (float*)carve((size_t)NN * H1D * 4);

  const int MBLK = (NN / 16 + 7) / 8;   // 8 M-tiles (waves) per block

  // relation weights (transposed f16) + input activation conversion
  build_wt<<<(RR*IND*H1D + 255)/256, 256, 0, stream>>>(basis1, comp1, wt1, IND, H1D);
  build_wt<<<(RR*H1D*H1D + 255)/256, 256, 0, stream>>>(basis2, comp2, wt2, H1D, H1D);
  build_wt<<<(RR*H1D*OUTD + 255)/256, 256, 0, stream>>>(basis3, comp3, wt3, H1D, OUTD);
  cvt_f16<<<(NN*IND + 255)/256, 256, 0, stream>>>(x, xf, NN*IND);

  // ---------------- layer 1: 128 -> 4x64 (concat 256), ReLU ----------------
  rgat_gemm_wmma<<<dim3(MBLK, RR, H1D/64), 256, 0, stream>>>(xf, wt1, xr, IND, H1D);
  (void)hipMemsetAsync(amax, 0xFF, (size_t)NN*4*4, stream);  // -NaN: bottom for atomicMaxF
  (void)hipMemsetAsync(denom, 0, (size_t)NN*4*4, stream);
  (void)hipMemsetAsync(agg, 0, (size_t)NN*H1D*4, stream);
  edge_logits<<<(EE*4 + 255)/256, 256, 0, stream>>>(xr, ei, et, q1, k1, alog, amax, 4, 64, H1D);
  edge_exp<<<(EE*4 + 255)/256, 256, 0, stream>>>(ei, alog, amax, denom, 4);
  edge_scatter<<<(EE*H1D + 255)/256, 256, 0, stream>>>(xr, ei, et, alog, denom, agg, 4, 64, H1D);
  finish_relu_f16<<<(NN*H1D + 255)/256, 256, 0, stream>>>(agg, bias1, xf, H1D, NN*H1D);

  // ---------------- layer 2: 256 -> 1x256, ReLU ----------------
  rgat_gemm_wmma<<<dim3(MBLK, RR, H1D/64), 256, 0, stream>>>(xf, wt2, xr, H1D, H1D);
  (void)hipMemsetAsync(amax, 0xFF, (size_t)NN*1*4, stream);
  (void)hipMemsetAsync(denom, 0, (size_t)NN*1*4, stream);
  (void)hipMemsetAsync(agg, 0, (size_t)NN*H1D*4, stream);
  edge_logits<<<(EE*1 + 255)/256, 256, 0, stream>>>(xr, ei, et, q2, k2, alog, amax, 1, H1D, H1D);
  edge_exp<<<(EE*1 + 255)/256, 256, 0, stream>>>(ei, alog, amax, denom, 1);
  edge_scatter<<<(EE*H1D + 255)/256, 256, 0, stream>>>(xr, ei, et, alog, denom, agg, 1, H1D, H1D);
  finish_relu_f16<<<(NN*H1D + 255)/256, 256, 0, stream>>>(agg, bias2, xf, H1D, NN*H1D);

  // ---------------- layer 3: 256 -> 1x64, mean(1 head)=identity, +bias -----
  rgat_gemm_wmma<<<dim3(MBLK, RR, OUTD/64), 256, 0, stream>>>(xf, wt3, xr, H1D, OUTD);
  (void)hipMemsetAsync(amax, 0xFF, (size_t)NN*1*4, stream);
  (void)hipMemsetAsync(denom, 0, (size_t)NN*1*4, stream);
  (void)hipMemsetAsync(agg, 0, (size_t)NN*OUTD*4, stream);
  edge_logits<<<(EE*1 + 255)/256, 256, 0, stream>>>(xr, ei, et, q3, k3, alog, amax, 1, OUTD, OUTD);
  edge_exp<<<(EE*1 + 255)/256, 256, 0, stream>>>(ei, alog, amax, denom, 1);
  edge_scatter<<<(EE*OUTD + 255)/256, 256, 0, stream>>>(xr, ei, et, alog, denom, agg, 1, OUTD, OUTD);
  finish_bias_f32<<<(NN*OUTD + 255)/256, 256, 0, stream>>>(agg, bias3, out, OUTD, NN*OUTD);
}
